// EnhancedGNNTransformerEncoder_10385230922200
// MI455X (gfx1250) — compile-verified, matrix operand order checked by reading, not run on hardware
//
#include <hip/hip_runtime.h>
#include <hip/hip_bf16.h>
#include <cstdint>
#include <cstddef>

// ---------------- problem constants (match reference) ----------------
#define NNODES 50000
#define NEDGES 800000
#define DIM    128      // D == H*C == OUT
#define NHEAD  8
#define CDIM   16
#define NLAYER 4
#define WPK_ELEMS (4 * 8 * 32 * 16)   // kb x ntile x lane x elem = 16384 bf16 (32 KB)

typedef __attribute__((ext_vector_type(16))) __bf16 v16bf;
typedef __attribute__((ext_vector_type(8)))  __bf16 v8bf;
typedef __attribute__((ext_vector_type(8)))  float  v8f;

// float -> bf16, round-to-nearest-even, pure bit manipulation
static __device__ __forceinline__ __bf16 f2bf(float f) {
  union { float f; unsigned u; } c; c.f = f;
  unsigned u = c.u + 0x7FFFu + ((c.u >> 16) & 1u);
  unsigned short h = (unsigned short)(u >> 16);
  __bf16 r;
  __builtin_memcpy(&r, &h, sizeof(r));
  return r;
}

// ---------------------------------------------------------------------
// Pack W[128x128] f32 into bf16 WMMA B-fragments:
// Wpk[((kb*8+nt)*32 + lane)*16 + e] = bf16(W[kk*128 + n])
//   kk = kb*32 + (e<8 ? e : e+8) + (lane>=16 ? 8 : 0),  n = nt*16 + (lane&15)
// ---------------------------------------------------------------------
__global__ __launch_bounds__(256)
void pack_weights(const float* __restrict__ W, __bf16* __restrict__ Wpk) {
  const int tid = blockIdx.x * blockDim.x + threadIdx.x;
  if (tid >= WPK_ELEMS) return;
  const int e  = tid & 15;
  const int l  = (tid >> 4) & 31;
  const int nt = (tid >> 9) & 7;
  const int kb = tid >> 12;
  const int kk = kb * 32 + ((e < 8) ? e : (e + 8)) + ((l >= 16) ? 8 : 0);
  const int n  = nt * 16 + (l & 15);
  Wpk[tid] = f2bf(W[(size_t)kk * DIM + n]);
}

// f32 -> bf16 row-major conversion (layer-0 input)
__global__ __launch_bounds__(256)
void cvt_bf16(const float* __restrict__ x, __bf16* __restrict__ y, int n) {
  int i = blockIdx.x * blockDim.x + threadIdx.x;
  if (i < n) y[i] = f2bf(x[i]);
}

// fused ReLU + f32->bf16 (produces next layer's GEMM input)
__global__ __launch_bounds__(256)
void relu_cvt_bf16(const float* __restrict__ x, __bf16* __restrict__ y, int n) {
  int i = blockIdx.x * blockDim.x + threadIdx.x;
  if (i < n) { float t = x[i]; y[i] = f2bf(t > 0.f ? t : 0.f); }
}

// ---------------------------------------------------------------------
// Y[M x 128] = Xbf[M x 128](bf16) @ W(packed bf16) + bias
// block = 256 threads = 8 waves; block covers 128 rows; wave = 16-row strip.
// Packed W staged once per block through LDS (shared by all 8 waves).
// Per K-step: batch-load all 8 B fragments (one DS clause), then issue the
// 8 WMMAs back-to-back so DS latency is paid once, not per WMMA.
// ---------------------------------------------------------------------
__global__ __launch_bounds__(256)
void gemm128_wmma(const __bf16* __restrict__ Xbf, const __bf16* __restrict__ Wpk,
                  const float* __restrict__ bias, float* __restrict__ Y, int M) {
  __shared__ __align__(32) __bf16 sW[WPK_ELEMS];   // 32 KB

  // cooperative global -> LDS copy of the packed weight fragments (b128 chunks)
  {
    const uint4* __restrict__ g = (const uint4*)Wpk;
    uint4* s = (uint4*)sW;
    #pragma unroll
    for (int i = 0; i < WPK_ELEMS / (8 * 256); ++i)        // 8 uint4 per thread
      s[threadIdx.x + 256 * i] = g[threadIdx.x + 256 * i];
  }
  __syncthreads();

  const int wave   = threadIdx.x >> 5;
  const int lane   = threadIdx.x & 31;
  const int laneLo = lane & 15;
  const int hiHalf = lane >> 4;
  const int row0   = blockIdx.x * 128 + wave * 16;

  v8f acc[8];
  #pragma unroll
  for (int i = 0; i < 8; ++i)
    #pragma unroll
    for (int r = 0; r < 8; ++r) acc[i][r] = 0.f;

  int mA = row0 + laneLo;
  if (mA >= M) mA = M - 1;                      // clamp; masked at store
  const __bf16* __restrict__ xrow = Xbf + (size_t)mA * DIM;
  const __bf16* __restrict__ swl  = sW + (size_t)lane * 16;

  #pragma unroll
  for (int kb = 0; kb < 4; ++kb) {              // K = 4 x 32
    // A fragment: ISA 16-bit 16x32 layout = two contiguous 8-element K runs
    const int kbase = kb * 32 + hiHalf * 8;
    union { v16bf v; v8bf h[2]; } a;
    a.h[0] = *(const v8bf*)(xrow + kbase);       // K = kbase .. kbase+7
    a.h[1] = *(const v8bf*)(xrow + kbase + 16);  // K = kbase+16 .. kbase+23

    // batch-load the 8 B fragments for this K-step (single DS clause)
    v16bf b[8];
    #pragma unroll
    for (int nt = 0; nt < 8; ++nt)
      b[nt] = *(const v16bf*)(swl + (size_t)((kb * 8) + nt) * (32 * 16));

    // dense WMMA burst
    #pragma unroll
    for (int nt = 0; nt < 8; ++nt)
      acc[nt] = __builtin_amdgcn_wmma_f32_16x16x32_bf16(
          /*neg_a=*/false, a.v, /*neg_b=*/false, b[nt],
          /*c_mod=*/(short)0, acc[nt], /*reuse_a=*/false, /*reuse_b=*/false);
  }

  // C/D layout: VGPR r, lane l -> M = r + 8*(l>=16), N = l&15
  #pragma unroll
  for (int nt = 0; nt < 8; ++nt) {
    const int n  = nt * 16 + laneLo;
    const float bn = bias[n];
    #pragma unroll
    for (int r = 0; r < 8; ++r) {
      const int m = row0 + r + hiHalf * 8;
      if (m < M) Y[(size_t)m * DIM + n] = acc[nt][r] + bn;
    }
  }
}

// ---------------------------------------------------------------------
// Edge-phase kernels (segment softmax over incoming edges per dst node)
// ---------------------------------------------------------------------
static __device__ __forceinline__ void atomicMaxF(float* addr, float val) {
  unsigned* ua = (unsigned*)addr;
  unsigned old = *ua;
  for (;;) {
    float of;
    __builtin_memcpy(&of, &old, 4);
    if (of >= val) break;
    unsigned nv;
    __builtin_memcpy(&nv, &val, 4);
    unsigned assumed = old;
    old = atomicCAS(ua, assumed, nv);
    if (old == assumed) break;
  }
}

__global__ __launch_bounds__(256)
void init_stats(float* __restrict__ maxb, float* __restrict__ sumb, int n) {
  int i = blockIdx.x * blockDim.x + threadIdx.x;
  if (i < n) { maxb[i] = -3.0e38f; sumb[i] = 0.f; }
}

__global__ __launch_bounds__(256)
void edge_score(const int* __restrict__ src, const int* __restrict__ dst,
                const float* __restrict__ q, const float* __restrict__ k,
                float* __restrict__ score, float* __restrict__ maxb) {
  int idx = blockIdx.x * blockDim.x + threadIdx.x;
  if (idx >= NEDGES * NHEAD) return;
  const int e = idx >> 3;
  const int h = idx & 7;
  const int s = src[e], d = dst[e];
  const float* __restrict__ qp = q + (size_t)d * DIM + h * CDIM;
  const float* __restrict__ kp = k + (size_t)s * DIM + h * CDIM;
  float acc = 0.f;
  #pragma unroll
  for (int c = 0; c < CDIM; ++c) acc += qp[c] * kp[c];
  acc *= 0.25f;                         // 1/sqrt(16)
  score[idx] = acc;
  atomicMaxF(maxb + (size_t)d * NHEAD + h, acc);
}

__global__ __launch_bounds__(256)
void edge_exp(const int* __restrict__ dst, const float* __restrict__ maxb,
              float* __restrict__ score, float* __restrict__ sumb) {
  int idx = blockIdx.x * blockDim.x + threadIdx.x;
  if (idx >= NEDGES * NHEAD) return;
  const int e = idx >> 3;
  const int h = idx & 7;
  const int d = dst[e];
  const float p = __expf(score[idx] - maxb[(size_t)d * NHEAD + h]);
  score[idx] = p;
  atomicAdd(sumb + (size_t)d * NHEAD + h, p);
}

__global__ __launch_bounds__(256)
void edge_aggregate(const int* __restrict__ src, const int* __restrict__ dst,
                    const float* __restrict__ v, const float* __restrict__ score,
                    const float* __restrict__ sumb, float* __restrict__ xout) {
  int idx = blockIdx.x * blockDim.x + threadIdx.x;   // NEDGES*DIM < 2^31
  if (idx >= NEDGES * DIM) return;
  const int e = idx >> 7;
  const int t = idx & 127;
  const int h = t >> 4;
  const int s = src[e], d = dst[e];
  const float alpha = score[(size_t)e * NHEAD + h] / sumb[(size_t)d * NHEAD + h];
  atomicAdd(xout + (size_t)d * DIM + t, alpha * v[(size_t)s * DIM + t]);
}

// ---------------------------------------------------------------------
extern "C" void kernel_launch(void* const* d_in, const int* in_sizes, int n_in,
                              void* d_out, int out_size, void* d_ws, size_t ws_size,
                              hipStream_t stream) {
  const float* x    = (const float*)d_in[0];
  const int*   eidx = (const int*)  d_in[1];
  // d_in[2] = batch (unused)
  const float* Wq   = (const float*)d_in[3];
  const float* bq   = (const float*)d_in[4];
  const float* Wk   = (const float*)d_in[5];
  const float* bk   = (const float*)d_in[6];
  const float* Wv   = (const float*)d_in[7];
  const float* bv   = (const float*)d_in[8];
  const float* Wsk  = (const float*)d_in[9];
  const float* bsk  = (const float*)d_in[10];
  const float* Wout = (const float*)d_in[11];
  const float* bout = (const float*)d_in[12];

  const int* src = eidx;            // edge_index[0]
  const int* dst = eidx + NEDGES;   // edge_index[1]

  // workspace carve-up (float units; all offsets 16B-aligned)
  float* ws = (float*)d_ws;
  const size_t ND = (size_t)NNODES * DIM;         // 6.4M
  const size_t NH = (size_t)NNODES * NHEAD;       // 0.4M
  float*  xn    = ws;                             // f32 node accumulator
  float*  q     = xn + ND;
  float*  k     = q + ND;
  float*  v     = k + ND;
  float*  score = v + ND;                          // NEDGES*NHEAD (== ND)
  float*  maxb  = score + (size_t)NEDGES * NHEAD;
  float*  sumb  = maxb + NH;
  __bf16* xbf   = (__bf16*)(sumb + NH);            // ND bf16
  __bf16* Wpk   = xbf + ND;                        // WPK_ELEMS bf16 (32 KB)
  (void)ws_size; (void)in_sizes; (void)n_in; (void)out_size;

  const dim3 bGemm(256);
  const dim3 gGemm((NNODES + 127) / 128);
  const int pkBlocks  = (WPK_ELEMS + 255) / 256;
  const int ehBlocks  = (NEDGES * NHEAD + 255) / 256;
  const int aggBlocks = (NEDGES * DIM + 255) / 256;
  const int nhBlocks  = ((int)NH + 255) / 256;
  const int ndBlocks  = ((int)ND + 255) / 256;

  // layer-0 input: f32 -> bf16 once
  cvt_bf16<<<ndBlocks, 256, 0, stream>>>(x, xbf, (int)ND);

  for (int l = 0; l < NLAYER; ++l) {
    const size_t wOff = (size_t)l * DIM * DIM;
    const size_t bOff = (size_t)l * DIM;

    pack_weights<<<pkBlocks, 256, 0, stream>>>(Wq + wOff, Wpk);
    gemm128_wmma<<<gGemm, bGemm, 0, stream>>>(xbf, Wpk, bq + bOff, q, NNODES);
    pack_weights<<<pkBlocks, 256, 0, stream>>>(Wk + wOff, Wpk);
    gemm128_wmma<<<gGemm, bGemm, 0, stream>>>(xbf, Wpk, bk + bOff, k, NNODES);
    pack_weights<<<pkBlocks, 256, 0, stream>>>(Wv + wOff, Wpk);
    gemm128_wmma<<<gGemm, bGemm, 0, stream>>>(xbf, Wpk, bv + bOff, v, NNODES);
    pack_weights<<<pkBlocks, 256, 0, stream>>>(Wsk + wOff, Wpk);
    gemm128_wmma<<<gGemm, bGemm, 0, stream>>>(xbf, Wpk, bsk + bOff, xn, NNODES);

    init_stats    <<<nhBlocks, 256, 0, stream>>>(maxb, sumb, (int)NH);
    edge_score    <<<ehBlocks, 256, 0, stream>>>(src, dst, q, k, score, maxb);
    edge_exp      <<<ehBlocks, 256, 0, stream>>>(dst, maxb, score, sumb);
    edge_aggregate<<<aggBlocks, 256, 0, stream>>>(src, dst, v, score, sumb, xn);

    // ReLU fused with bf16 conversion -> next layer's GEMM input
    relu_cvt_bf16 <<<ndBlocks, 256, 0, stream>>>(xn, xbf, (int)ND);
  }

  pack_weights<<<pkBlocks, 256, 0, stream>>>(Wout, Wpk);
  gemm128_wmma<<<gGemm, bGemm, 0, stream>>>(xbf, Wpk, bout, (float*)d_out, NNODES);
}
